// get_model_39032662786230
// MI455X (gfx1250) — compile-verified, matrix operand order checked by reading, not run on hardware
//
#include <hip/hip_runtime.h>
#include <math.h>

// ---------------------------------------------------------------------------
// VN-DGCNN partseg forward for MI455X (gfx1250, wave32, WMMA).
// All channel contractions run through one WMMA GEMM kernel
// (v_wmma_f32_16x16x32_f16, f32 accumulate). Activations live in f16 with
// channel dims padded to multiples of 32, so each lane's A/B half-fragment is
// a single global_load_b128. Each wave carries NT output tiles so the A
// fragment is reused NT times (loads/wmma = 2 + 2/NT b128). Weights are
// zero-padded to [OpR(x16*NT), Cp(x32)] f16 per layer, which annihilates any
// garbage in A's pad columns. Workspace: ~315 MB.
// ---------------------------------------------------------------------------

typedef __attribute__((ext_vector_type(16))) _Float16 v16h;
typedef __attribute__((ext_vector_type(8)))  _Float16 v8h;
typedef __attribute__((ext_vector_type(8)))  float    v8f;

typedef _Float16 h16;

static constexpr int   Bb  = 4;
static constexpr int   Nn  = 2048;
static constexpr int   Kk  = 40;
static constexpr long  NB  = (long)Bb * Nn;   // 8192 point locations
static constexpr long  LsK = NB * Kk;         // 327680 edge locations
static constexpr long  R3  = NB * 3;          // rows per point-level GEMM

#define EPSv   1e-6f
#define BNEPS  1e-5f
#define SLOPEv 0.2f

// ------------------------------- GEMM (WMMA) -------------------------------
// Y[r, ycoff+o] = sum_c A[r*ars + c] * Wp[o*wrs + c]
// A: f16 rows readable up to Cp; Wp: f16 zero-padded [OpR, Cp], OpR = grid.y*16*NT.
// Per k-step: 2 b128 A loads shared across NT tiles, 2*NT b128 B loads, NT wmmas.
template <int NT>
__global__ __launch_bounds__(32)
void wmma_gemm_f16(const h16* __restrict__ A, long ars,
                   const h16* __restrict__ Wp, long wrs,
                   void* __restrict__ Yv, long yrs, long ycoff,
                   long L, int Cp, int O, int storeF32)
{
    const int  lane   = threadIdx.x;
    const int  half   = lane >> 4;
    const int  mn     = lane & 15;
    const long tileM  = blockIdx.x;
    const int  tileN0 = blockIdx.y * NT;

    long row = tileM * 16 + mn;
    if (row >= L) row = L - 1;                    // clamp; masked on store
    const h16* Ap = A + row * ars + half * 8;
    const h16* Bp = Wp + (long)(tileN0 * 16 + mn) * wrs + half * 8;

    v8f acc[NT];
#pragma unroll
    for (int nt = 0; nt < NT; ++nt) acc[nt] = (v8f){};

#pragma unroll 2
    for (int k0 = 0; k0 < Cp; k0 += 32) {
        __builtin_prefetch(Ap + k0 + 64, 0, 1);   // global_prefetch_b8
        v8h al = *(const v8h*)(Ap + k0);
        v8h ah = *(const v8h*)(Ap + k0 + 16);
        v16h af = __builtin_shufflevector(al, ah, 0, 1, 2, 3, 4, 5, 6, 7,
                                          8, 9, 10, 11, 12, 13, 14, 15);
#pragma unroll
        for (int nt = 0; nt < NT; ++nt) {
            const h16* bp = Bp + (long)nt * 16 * wrs + k0;
            v8h bl = *(const v8h*)bp;
            v8h bh = *(const v8h*)(bp + 16);
            v16h bf = __builtin_shufflevector(bl, bh, 0, 1, 2, 3, 4, 5, 6, 7,
                                              8, 9, 10, 11, 12, 13, 14, 15);
            acc[nt] = __builtin_amdgcn_wmma_f32_16x16x32_f16(
                false, af, false, bf, (short)0, acc[nt], false, false);
        }
    }

#pragma unroll
    for (int nt = 0; nt < NT; ++nt) {
        const int ocol = (tileN0 + nt) * 16 + mn;
        if (ocol < O) {
            if (storeF32) {
                float* Y = reinterpret_cast<float*>(Yv);
#pragma unroll
                for (int v = 0; v < 8; ++v) {     // D tile: M = v + 8*half
                    long r = tileM * 16 + v + 8 * half;
                    if (r < L) Y[r * yrs + ycoff + ocol] = acc[nt][v];
                }
            } else {
                h16* Y = reinterpret_cast<h16*>(Yv);
#pragma unroll
                for (int v = 0; v < 8; ++v) {
                    long r = tileM * 16 + v + 8 * half;
                    if (r < L) Y[r * yrs + ycoff + ocol] = (h16)acc[nt][v];
                }
            }
        }
    }
}

// weight f32 [O,C] -> f16 zero-padded [Op, Cp]
__global__ void wcvt_kernel(const float* __restrict__ W, h16* __restrict__ Wp,
                            int O, int C, int Op, int Cp)
{
    long t = blockIdx.x * (long)blockDim.x + threadIdx.x;
    if (t >= (long)Op * Cp) return;
    int c = (int)(t % Cp), o = (int)(t / Cp);
    Wp[t] = (o < O && c < C) ? (h16)W[(long)o * C + c] : (h16)0.f;
}

// complex weight concat: Wp[192,352] = [Wr | -Wi] zero-padded
__global__ void wcat_kernel(const float* __restrict__ Wr,
                            const float* __restrict__ Wi,
                            h16* __restrict__ Wp)
{
    long t = blockIdx.x * (long)blockDim.x + threadIdx.x;
    if (t >= 192L * 352) return;
    int c = (int)(t % 352), o = (int)(t / 352);
    float v = 0.f;
    if (o < 170) {
        if (c < 170)      v = Wr[o * 170 + c];
        else if (c < 340) v = -Wi[o * 170 + (c - 170)];
    }
    Wp[t] = (h16)v;
}

// x [B,3,N] f32 -> pts [B,N,3] f16
__global__ void xin_transpose_kernel(const float* __restrict__ x,
                                     h16* __restrict__ pts)
{
    long t = blockIdx.x * (long)blockDim.x + threadIdx.x;
    if (t >= (long)Bb * 3 * Nn) return;
    long n = t % Nn;
    long v = (t / Nn) % 3;
    long b = t / ((long)Nn * 3);
    pts[(b * Nn + n) * 3 + v] = (h16)x[(b * 3 + v) * Nn + n];
}

// --------------------------------- KNN ------------------------------------
// score = -||p-q||^2; top-40 descending, ties -> lower index (lax.top_k).
__global__ void knn_kernel(const h16* __restrict__ pts, long pstride,
                           long vstride, int C, int* __restrict__ idx)
{
    int b = blockIdx.y;
    int n = blockIdx.x * blockDim.x + threadIdx.x;
    if (n >= Nn) return;
    const h16* base = pts + (long)b * Nn * pstride;
    float q[64];
    {
        const h16* qp = base + (long)n * pstride;
        for (int v = 0; v < 3; ++v)
            for (int c = 0; c < C; ++c) q[v * C + c] = (float)qp[v * vstride + c];
    }
    float bv[Kk]; int bi[Kk];
    for (int k = 0; k < Kk; ++k) { bv[k] = -3.4e38f; bi[k] = 0; }
    for (int m = 0; m < Nn; ++m) {
        const h16* p = base + (long)m * pstride;
        float d2 = 0.f;
        for (int v = 0; v < 3; ++v)
            for (int c = 0; c < C; ++c) {
                float t = q[v * C + c] - (float)p[v * vstride + c];
                d2 = fmaf(t, t, d2);
            }
        float sc = -d2;
        if (sc > bv[Kk - 1]) {
            int k = Kk - 1;
            while (k > 0 && bv[k - 1] < sc) {
                bv[k] = bv[k - 1]; bi[k] = bi[k - 1]; --k;
            }
            bv[k] = sc; bi[k] = m;
        }
    }
    int* o = idx + ((long)b * Nn + n) * Kk;
    for (int k = 0; k < Kk; ++k) o[k] = bi[k];
}

// feat [B,N,K,3,Cp]: c<C -> pts[j]-pts[n]; c<2C -> pts[n]; else 0
__global__ void edge_kernel(const h16* __restrict__ pts, long pstride,
                            long vstride, int C, const int* __restrict__ idx,
                            h16* __restrict__ feat, int Cp, long total)
{
    long t = blockIdx.x * (long)blockDim.x + threadIdx.x;
    if (t >= total) return;
    long r = t;
    int  c2 = (int)(r % Cp); r /= Cp;
    int  v  = (int)(r % 3);  r /= 3;
    int  k  = (int)(r % Kk); r /= Kk;
    long n  = r % Nn;
    long b  = r / Nn;
    if (c2 >= 2 * C) { feat[t] = (h16)0.f; return; }
    long pb = b * Nn * pstride;
    int  cc = (c2 < C) ? c2 : c2 - C;
    float ctr = (float)pts[pb + n * pstride + v * vstride + cc];
    float val;
    if (c2 < C) {
        int j = idx[(b * Nn + n) * Kk + k];
        val = (float)pts[pb + (long)j * pstride + v * vstride + cc] - ctr;
    } else {
        val = ctr;
    }
    feat[t] = (h16)val;
}

// ------------------------------- BN statistics ----------------------------
// vecdim==3: value = ||Y[s,*,c]|| + EPS; vecdim==1: scalar Y[s,c].
__global__ __launch_bounds__(256)
void stats_kernel(const h16* __restrict__ Y, long rstride, long coff,
                  float* __restrict__ out, long S, int C, int vecdim)
{
    int  c    = blockIdx.y;
    long s0   = blockIdx.x * 256L + threadIdx.x;
    long step = (long)gridDim.x * 256L;
    float sum = 0.f, sq = 0.f;
    for (long s = s0; s < S; s += step) {
        float val;
        if (vecdim == 3) {
            long base = s * 3;
            float a0 = (float)Y[(base + 0) * rstride + coff + c];
            float a1 = (float)Y[(base + 1) * rstride + coff + c];
            float a2 = (float)Y[(base + 2) * rstride + coff + c];
            val = sqrtf(a0 * a0 + a1 * a1 + a2 * a2) + EPSv;
        } else {
            val = (float)Y[s * rstride + coff + c];
        }
        sum += val; sq += val * val;
    }
    __shared__ float sh0[256];
    __shared__ float sh1[256];
    sh0[threadIdx.x] = sum; sh1[threadIdx.x] = sq;
    __syncthreads();
    for (int w = 128; w > 0; w >>= 1) {
        if ((int)threadIdx.x < w) {
            sh0[threadIdx.x] += sh0[threadIdx.x + w];
            sh1[threadIdx.x] += sh1[threadIdx.x + w];
        }
        __syncthreads();
    }
    if (threadIdx.x == 0) {
        atomicAdd(&out[c], sh0[0]);
        atomicAdd(&out[C + c], sh1[0]);
    }
}

// -------------------------- VN BN + VN LeakyReLU ---------------------------
__global__ void vn_apply_kernel(const h16* __restrict__ Yw, long yrs, long ycoff,
                                const h16* __restrict__ Dv, long drs, long dcoff,
                                int dC,
                                h16* __restrict__ Out, long ors, long ocoff,
                                const float* __restrict__ stats,
                                const float* __restrict__ gam,
                                const float* __restrict__ bet,
                                float invS, long S, int C)
{
    long t = blockIdx.x * (long)blockDim.x + threadIdx.x;
    if (t >= S * (long)C) return;
    int  c = (int)(t % C);
    long s = t / C;
    float p0 = (float)Yw[(s * 3 + 0) * yrs + ycoff + c];
    float p1 = (float)Yw[(s * 3 + 1) * yrs + ycoff + c];
    float p2 = (float)Yw[(s * 3 + 2) * yrs + ycoff + c];
    if (stats) {
        float nn   = sqrtf(p0 * p0 + p1 * p1 + p2 * p2) + EPSv;
        float mean = stats[c] * invS;
        float var  = stats[C + c] * invS - mean * mean;
        float nbn  = gam[c] * (nn - mean) * rsqrtf(var + BNEPS) + bet[c];
        float scl  = nbn / nn;
        p0 *= scl; p1 *= scl; p2 *= scl;
    }
    int cd = (dC == 1) ? 0 : c;
    float d0 = (float)Dv[(s * 3 + 0) * drs + dcoff + cd];
    float d1 = (float)Dv[(s * 3 + 1) * drs + dcoff + cd];
    float d2 = (float)Dv[(s * 3 + 2) * drs + dcoff + cd];
    float dot = p0 * d0 + p1 * d1 + p2 * d2;
    float dsq = d0 * d0 + d1 * d1 + d2 * d2;
    float coef = (dot >= 0.f) ? 0.f : (1.f - SLOPEv) * dot / (dsq + EPSv);
    Out[(s * 3 + 0) * ors + ocoff + c] = (h16)(p0 - coef * d0);
    Out[(s * 3 + 1) * ors + ocoff + c] = (h16)(p1 - coef * d1);
    Out[(s * 3 + 2) * ors + ocoff + c] = (h16)(p2 - coef * d2);
}

// ------------------------------- VN max pool -------------------------------
// X, Dv: [B*N, K, 3, rs]; argmax_k <x,d> (first max), write selected vector.
__global__ void pool_kernel(const h16* __restrict__ X,
                            const h16* __restrict__ Dv, long rs,
                            h16* __restrict__ Out, long ors, long ocoff, int C)
{
    long t = blockIdx.x * (long)blockDim.x + threadIdx.x;
    if (t >= NB * (long)C) return;
    int  c = (int)(t % C);
    long s = t / C;
    long base = (s * Kk) * 3L * rs;
    float best = -3.4e38f; int bk = 0;
    for (int k = 0; k < Kk; ++k) {
        long o = base + (long)k * 3 * rs + c;
        float dot = (float)X[o] * (float)Dv[o] +
                    (float)X[o + rs] * (float)Dv[o + rs] +
                    (float)X[o + 2 * rs] * (float)Dv[o + 2 * rs];
        if (dot > best) { best = dot; bk = k; }
    }
    long o = base + (long)bk * 3 * rs + c;
    Out[(s * 3 + 0) * ors + ocoff + c] = X[o];
    Out[(s * 3 + 1) * ors + ocoff + c] = X[o + rs];
    Out[(s * 3 + 2) * ors + ocoff + c] = X[o + 2 * rs];
}

// comb [B*N,3,352] (cols 0..169 = x); J [B*N,3,192]; write x_im -> cols 170..339
__global__ void crossj_kernel(h16* __restrict__ comb, const h16* __restrict__ J)
{
    long t = blockIdx.x * (long)blockDim.x + threadIdx.x;
    if (t >= NB * 170L) return;
    int  c = (int)(t % 170);
    long s = t / 170;
    float jx = (float)J[(s * 3 + 0) * 192 + c];
    float jy = (float)J[(s * 3 + 1) * 192 + c];
    float jz = (float)J[(s * 3 + 2) * 192 + c];
    float rn = 1.f / (sqrtf(jx * jx + jy * jy + jz * jz) + EPSv);
    jx *= rn; jy *= rn; jz *= rn;
    float x0 = (float)comb[(s * 3 + 0) * 352 + c];
    float x1 = (float)comb[(s * 3 + 1) * 352 + c];
    float x2 = (float)comb[(s * 3 + 2) * 352 + c];
    comb[(s * 3 + 0) * 352 + 170 + c] = (h16)(jy * x2 - jz * x1);
    comb[(s * 3 + 1) * 352 + 170 + c] = (h16)(jz * x0 - jx * x2);
    comb[(s * 3 + 2) * 352 + 170 + c] = (h16)(jx * x1 - jy * x0);
}

// H [B,N,3,704]: cols 341..681 = mean over N of cols 0..340
__global__ void meanfill_kernel(h16* __restrict__ H)
{
    int t = blockIdx.x * blockDim.x + threadIdx.x;
    if (t >= Bb * 3 * 341) return;
    int c = t % 341;
    int v = (t / 341) % 3;
    int b = t / (341 * 3);
    float s = 0.f;
    for (long n = 0; n < Nn; ++n)
        s += (float)H[(((long)b * Nn + n) * 3 + v) * 704 + c];
    float m = s / (float)Nn;
    for (long n = 0; n < Nn; ++n)
        H[(((long)b * Nn + n) * 3 + v) * 704 + 341 + c] = (h16)m;
}

// Out[s, ocoff + i*3+k] = sum_j H[s,j,i] * z0[s,k,j]   (H stride hs, z0 [.,3,3])
__global__ void stdfeat_kernel(const h16* __restrict__ H, long hs, int CH,
                               const h16* __restrict__ z0,
                               h16* __restrict__ Out, long ors, long ocoff)
{
    long t = blockIdx.x * (long)blockDim.x + threadIdx.x;
    if (t >= NB * (long)CH) return;
    int  i = (int)(t % CH);
    long s = t / CH;
    float h0 = (float)H[(s * 3 + 0) * hs + i];
    float h1 = (float)H[(s * 3 + 1) * hs + i];
    float h2 = (float)H[(s * 3 + 2) * hs + i];
#pragma unroll
    for (int k = 0; k < 3; ++k) {
        const h16* z = &z0[(s * 3 + k) * 3];
        Out[s * ors + ocoff + (long)i * 3 + k] =
            (h16)(h0 * (float)z[0] + h1 * (float)z[1] + h2 * (float)z[2]);
    }
}

// hg[b,c] = max over n of X[b,n,c]  (C = 2046)
__global__ void colmax_kernel(const h16* __restrict__ X, float* __restrict__ out)
{
    int t = blockIdx.x * blockDim.x + threadIdx.x;
    if (t >= Bb * 2046) return;
    int c = t % 2046, b = t / 2046;
    float m = -3.4e38f;
    for (long n = 0; n < Nn; ++n)
        m = fmaxf(m, (float)X[((long)b * Nn + n) * 2046 + c]);
    out[(long)b * 2046 + c] = m;
}

// label branch: lf[b,o] = lrelu(bn1d(c7W @ l))  (stats over batch only)
__global__ void label_kernel(const float* __restrict__ l,
                             const float* __restrict__ Wm,
                             const float* __restrict__ g,
                             const float* __restrict__ bb,
                             float* __restrict__ lf)
{
    int o = threadIdx.x;
    if (o >= 64) return;
    float y[Bb];
    float sum = 0.f, sq = 0.f;
    for (int b = 0; b < Bb; ++b) {
        float acc = 0.f;
        for (int c = 0; c < 16; ++c) acc += Wm[o * 16 + c] * l[b * 16 + c];
        y[b] = acc; sum += acc; sq += acc * acc;
    }
    float mean = sum / (float)Bb;
    float var  = sq / (float)Bb - mean * mean;
    float is   = rsqrtf(var + BNEPS);
    for (int b = 0; b < Bb; ++b) {
        float t = g[o] * (y[b] - mean) * is + bb[o];
        lf[b * 64 + o] = (t >= 0.f) ? t : SLOPEv * t;
    }
}

// Hc[b,n,c<2110] = c<2046 ? hg[b,c] : lf[b,c-2046]    (Hc stride 2304)
__global__ void fillg_kernel(h16* __restrict__ Hc, const float* __restrict__ hg,
                             const float* __restrict__ lf)
{
    long t = blockIdx.x * (long)blockDim.x + threadIdx.x;
    if (t >= NB * 2110L) return;
    int  c = (int)(t % 2110);
    long s = t / 2110;
    long b = s / Nn;
    float v = (c < 2046) ? hg[b * 2046 + c] : lf[b * 64 + (c - 2046)];
    Hc[s * 2304 + c] = (h16)v;
}

// scalar BN + LeakyReLU, in place (f16)
__global__ void bn_lrelu_kernel(h16* __restrict__ Y,
                                const float* __restrict__ stats,
                                const float* __restrict__ g,
                                const float* __restrict__ bb,
                                float invL, long total, int C)
{
    long t = blockIdx.x * (long)blockDim.x + threadIdx.x;
    if (t >= total) return;
    int c = (int)(t % C);
    float mean = stats[c] * invL;
    float var  = stats[C + c] * invL - mean * mean;
    float y = g[c] * ((float)Y[t] - mean) * rsqrtf(var + BNEPS) + bb[c];
    Y[t] = (h16)((y >= 0.f) ? y : SLOPEv * y);
}

// ---------------------------------------------------------------------------
// Host side
// ---------------------------------------------------------------------------

enum {
    IN_X = 0, IN_L,
    P_c1W, P_c1D, P_c2W, P_c2D, P_c3W, P_c3D, P_c4W, P_c4D, P_c5W, P_c5D,
    P_p1D, P_p2D, P_p3D, P_l1W, P_l2W, P_l3W, P_l4W,
    P_cl1Wr, P_cl1Wi, P_cl1D, P_cl2Wr, P_cl2Wi, P_cl2D,
    P_c6W, P_c6D, P_s1W, P_s1D, P_s2W, P_s2D, P_s3W,
    P_c7W, P_c8W, P_c9W, P_c10W, P_c11W,
    P_c1g, P_c1b, P_c2g, P_c2b, P_c3g, P_c3b, P_c4g, P_c4b, P_c5g, P_c5b,
    P_c6g, P_c6b, P_s1g, P_s1b, P_s2g, P_s2b, P_c7g, P_c7b,
    P_c8g, P_c8b, P_c9g, P_c9b, P_c10g, P_c10b
};

// workspace layout (bytes, f16 activations). Early arena re-aliased late.
static constexpr size_t SZ_FEAT  = (size_t)LsK * 3 * 64 * 2;   // 125,829,120
static constexpr size_t SZ_H     = (size_t)LsK * 3 * 32 * 2;   //  62,914,560
static constexpr size_t OFF_FEAT = 0;
static constexpr size_t OFF_BUFA = SZ_FEAT;
static constexpr size_t OFF_BUFB = OFF_BUFA + SZ_H;
static constexpr size_t ARENA_END = OFF_BUFB + SZ_H;           // 251,658,240
static constexpr size_t OFF_BUFC1 = OFF_FEAT;
static constexpr size_t OFF_BUFC2 = OFF_FEAT + SZ_H;
// late-stage aliases inside the arena (valid once pooling stages are done)
static constexpr size_t OFF_COMB2 = 0;
static constexpr size_t OFF_H682  = OFF_COMB2 + (size_t)R3 * 352 * 2;
static constexpr size_t OFF_Z1    = OFF_H682  + (size_t)R3 * 704 * 2;
static constexpr size_t OFF_Z2    = OFF_Z1    + (size_t)R3 * 352 * 2;
static constexpr size_t OFF_Z0    = OFF_Z2    + (size_t)R3 * 192 * 2;
static constexpr size_t OFF_HSTD  = OFF_Z0    + (size_t)R3 * 3 * 2 + 64;
static constexpr size_t OFF_HC    = OFF_HSTD  + (size_t)NB * 2046 * 2;
// persistent buffers after the arena
static constexpr size_t OFF_PTS0 = ARENA_END;
static constexpr size_t OFF_IDX  = OFF_PTS0 + (size_t)R3 * 2 + 128;
static constexpr size_t OFF_X123 = OFF_IDX  + (size_t)NB * Kk * 4;
static constexpr size_t OFF_COMB = OFF_X123 + (size_t)R3 * 64 * 2;
static constexpr size_t OFF_JA   = OFF_COMB + (size_t)R3 * 352 * 2;
static constexpr size_t OFF_RE   = OFF_JA   + (size_t)R3 * 192 * 2;
static constexpr size_t OFF_DCL  = OFF_RE   + (size_t)R3 * 192 * 2;
static constexpr size_t OFF_D6   = OFF_DCL  + (size_t)R3 * 192 * 2;
static constexpr size_t OFF_HG   = OFF_D6   + (size_t)R3 * 2 + 128;
static constexpr size_t OFF_LF   = OFF_HG   + (size_t)Bb * 2046 * 4 + 64;
static constexpr size_t OFF_Y8   = OFF_LF   + (size_t)Bb * 64 * 4;
static constexpr size_t OFF_Y9   = OFF_Y8   + (size_t)NB * 256 * 2;
static constexpr size_t OFF_Y10  = OFF_Y9   + (size_t)NB * 256 * 2;
static constexpr size_t OFF_STAT = OFF_Y10  + (size_t)NB * 128 * 2;
static constexpr size_t OFF_WPAD = OFF_STAT + 8192;            // 2 MB pad buf

static inline h16* WSH(void* ws, size_t off) {
    return reinterpret_cast<h16*>(reinterpret_cast<char*>(ws) + off);
}
static inline float* WSF(void* ws, size_t off) {
    return reinterpret_cast<float*>(reinterpret_cast<char*>(ws) + off);
}

static inline unsigned blk(long total) { return (unsigned)((total + 255) / 256); }

// raw gemm on already-padded f16 weights; OpR must be grid-covered (x16*NT)
static void gemm_raw(hipStream_t st, const h16* A, long ars,
                     const h16* Wp, long wrs, void* Y, long yrs, long ycoff,
                     long L, int Cp, int O, int OpR, int NT, int storeF32)
{
    dim3 grid((unsigned)((L + 15) / 16), (unsigned)(OpR / (16 * NT)), 1);
    if (NT == 1)
        wmma_gemm_f16<1><<<grid, dim3(32, 1, 1), 0, st>>>(A, ars, Wp, wrs, Y,
                                                          yrs, ycoff, L, Cp, O,
                                                          storeF32);
    else if (NT == 2)
        wmma_gemm_f16<2><<<grid, dim3(32, 1, 1), 0, st>>>(A, ars, Wp, wrs, Y,
                                                          yrs, ycoff, L, Cp, O,
                                                          storeF32);
    else
        wmma_gemm_f16<4><<<grid, dim3(32, 1, 1), 0, st>>>(A, ars, Wp, wrs, Y,
                                                          yrs, ycoff, L, Cp, O,
                                                          storeF32);
}

// convert f32 weight then gemm; picks NT per layer width
static void gemm_c(hipStream_t st, const h16* A, long ars, const float* W32,
                   h16* Wpad, void* Y, long yrs, long ycoff,
                   long L, int C, int O, int storeF32)
{
    int Cp = (C + 31) & ~31;
    int Op = (O + 15) & ~15;
    int NT = (Op <= 16) ? 1 : ((Op <= 32) ? 2 : 4);
    int OpR = (Op + 16 * NT - 1) & ~(16 * NT - 1);
    wcvt_kernel<<<blk((long)OpR * Cp), 256, 0, st>>>(W32, Wpad, O, C, OpR, Cp);
    gemm_raw(st, A, ars, Wpad, Cp, Y, yrs, ycoff, L, Cp, O, OpR, NT, storeF32);
}

static void bn_stats(hipStream_t st, const h16* Y, long rstride, long coff,
                     float* statsb, long S, int C, int vecdim)
{
    hipMemsetAsync(statsb, 0, sizeof(float) * 2 * C, st);
    unsigned gx = (unsigned)((S + 255) / 256);
    if (gx > 256) gx = 256;
    stats_kernel<<<dim3(gx, (unsigned)C), dim3(256), 0, st>>>(
        Y, rstride, coff, statsb, S, C, vecdim);
}

// one VN conv layer (Cout = 21, stride-32 buffers)
static void vn_conv(hipStream_t st, const h16* A, long ars, long S,
                    int Cin, const float* Wm, const float* Dm,
                    const float* g, const float* b,
                    h16* Yw, h16* Yd, float* statsb, h16* Wpad,
                    h16* Out, long ors, long ocoff)
{
    long L = S * 3;
    gemm_c(st, A, ars, Wm, Wpad, Yw, 32, 0, L, Cin, 21, 0);
    gemm_c(st, A, ars, Dm, Wpad, Yd, 32, 0, L, Cin, 21, 0);
    bn_stats(st, Yw, 32, 0, statsb, S, 21, 3);
    vn_apply_kernel<<<blk(S * 21), 256, 0, st>>>(
        Yw, 32, 0, Yd, 32, 0, 21, Out, ors, ocoff,
        statsb, g, b, 1.f / (float)S, S, 21);
}

extern "C" void kernel_launch(void* const* d_in, const int* in_sizes, int n_in,
                              void* d_out, int out_size, void* d_ws,
                              size_t ws_size, hipStream_t stream)
{
    (void)in_sizes; (void)n_in; (void)out_size; (void)ws_size;
    auto P = [&](int i) { return reinterpret_cast<const float*>(d_in[i]); };

    h16*  feat  = WSH(d_ws, OFF_FEAT);
    h16*  bufA  = WSH(d_ws, OFF_BUFA);
    h16*  bufB  = WSH(d_ws, OFF_BUFB);
    h16*  bufC1 = WSH(d_ws, OFF_BUFC1);
    h16*  bufC2 = WSH(d_ws, OFF_BUFC2);
    h16*  pts0  = WSH(d_ws, OFF_PTS0);
    int*  idx   = reinterpret_cast<int*>(WSF(d_ws, OFF_IDX));
    h16*  x123  = WSH(d_ws, OFF_X123);
    h16*  comb  = WSH(d_ws, OFF_COMB);
    h16*  ja    = WSH(d_ws, OFF_JA);
    h16*  re    = WSH(d_ws, OFF_RE);
    h16*  dcl   = WSH(d_ws, OFF_DCL);
    h16*  d6    = WSH(d_ws, OFF_D6);
    float* hg   = WSF(d_ws, OFF_HG);
    float* lf   = WSF(d_ws, OFF_LF);
    h16*  Y8    = WSH(d_ws, OFF_Y8);
    h16*  Y9    = WSH(d_ws, OFF_Y9);
    h16*  Y10   = WSH(d_ws, OFF_Y10);
    float* statsb = WSF(d_ws, OFF_STAT);
    h16*  Wpad  = WSH(d_ws, OFF_WPAD);
    h16*  comb2 = WSH(d_ws, OFF_COMB2);
    h16*  h682  = WSH(d_ws, OFF_H682);
    h16*  z1    = WSH(d_ws, OFF_Z1);
    h16*  z2    = WSH(d_ws, OFF_Z2);
    h16*  z0    = WSH(d_ws, OFF_Z0);
    h16*  hstd  = WSH(d_ws, OFF_HSTD);
    h16*  Hc    = WSH(d_ws, OFF_HC);
    h16*  zd    = hstd;  // s1/s2 direction buffer, reused before hstd is written

    const float* xin = P(IN_X);
    const float* lab = P(IN_L);
    float* outp = reinterpret_cast<float*>(d_out);

    // ---- stage 1: graph feature from raw points (C=1, Cp=32) ----
    xin_transpose_kernel<<<blk(R3), 256, 0, stream>>>(xin, pts0);
    knn_kernel<<<dim3((Nn + 127) / 128, Bb), 128, 0, stream>>>(pts0, 3, 1, 1, idx);
    edge_kernel<<<blk(LsK * 3 * 32), 256, 0, stream>>>(pts0, 3, 1, 1, idx, feat,
                                                       32, LsK * 3 * 32);
    vn_conv(stream, feat, 32, LsK, 2, P(P_c1W), P(P_c1D), P(P_c1g), P(P_c1b),
            bufA, bufB, statsb, Wpad, bufA, 32, 0);
    vn_conv(stream, bufA, 32, LsK, 21, P(P_c2W), P(P_c2D), P(P_c2g), P(P_c2b),
            bufC1, bufC2, statsb, Wpad, bufC1, 32, 0);
    gemm_c(stream, bufC1, 32, P(P_p1D), Wpad, bufB, 32, 0, LsK * 3, 21, 21, 0);
    pool_kernel<<<blk(NB * 21), 256, 0, stream>>>(bufC1, bufB, 32, x123, 64, 0, 21);

    // ---- stage 2 (Cp=64) ----
    knn_kernel<<<dim3((Nn + 127) / 128, Bb), 128, 0, stream>>>(x123, 192, 64, 21, idx);
    edge_kernel<<<blk(LsK * 3 * 64), 256, 0, stream>>>(x123, 192, 64, 21, idx,
                                                       feat, 64, LsK * 3 * 64);
    vn_conv(stream, feat, 64, LsK, 42, P(P_c3W), P(P_c3D), P(P_c3g), P(P_c3b),
            bufA, bufB, statsb, Wpad, bufA, 32, 0);
    vn_conv(stream, bufA, 32, LsK, 21, P(P_c4W), P(P_c4D), P(P_c4g), P(P_c4b),
            bufC1, bufC2, statsb, Wpad, bufC1, 32, 0);
    gemm_c(stream, bufC1, 32, P(P_p2D), Wpad, bufB, 32, 0, LsK * 3, 21, 21, 0);
    pool_kernel<<<blk(NB * 21), 256, 0, stream>>>(bufC1, bufB, 32, x123, 64, 21, 21);

    // ---- stage 3 ----
    knn_kernel<<<dim3((Nn + 127) / 128, Bb), 128, 0, stream>>>(x123 + 21, 192, 64, 21, idx);
    edge_kernel<<<blk(LsK * 3 * 64), 256, 0, stream>>>(x123 + 21, 192, 64, 21, idx,
                                                       feat, 64, LsK * 3 * 64);
    vn_conv(stream, feat, 64, LsK, 42, P(P_c5W), P(P_c5D), P(P_c5g), P(P_c5b),
            bufA, bufB, statsb, Wpad, bufA, 32, 0);
    gemm_c(stream, bufA, 32, P(P_p3D), Wpad, bufC1, 32, 0, LsK * 3, 21, 21, 0);
    pool_kernel<<<blk(NB * 21), 256, 0, stream>>>(bufA, bufC1, 32, x123, 64, 42, 21);

    // ---- complex linear block 1 ----
    gemm_c(stream, x123, 64, P(P_l1W), Wpad, comb, 352, 0, R3, 63, 170, 0);
    gemm_c(stream, x123, 64, P(P_l2W), Wpad, ja, 192, 0, R3, 63, 170, 0);
    crossj_kernel<<<blk(NB * 170), 256, 0, stream>>>(comb, ja);
    wcat_kernel<<<blk(192L * 352), 256, 0, stream>>>(P(P_cl1Wr), P(P_cl1Wi), Wpad);
    gemm_raw(stream, comb, 352, Wpad, 352, re, 192, 0, R3, 352, 170, 192, 4, 0);
    gemm_c(stream, comb, 352, P(P_cl1D), Wpad, dcl, 192, 0, R3, 170, 170, 0);
    vn_apply_kernel<<<blk(NB * 170), 256, 0, stream>>>(
        re, 192, 0, dcl, 192, 0, 170, comb, 352, 170,
        (const float*)nullptr, (const float*)nullptr, (const float*)nullptr,
        0.f, NB, 170);

    // ---- complex linear block 2 ----
    gemm_c(stream, comb, 352, P(P_l3W), Wpad, comb2, 352, 0, R3, 340, 170, 0);
    gemm_c(stream, comb, 352, P(P_l4W), Wpad, ja, 192, 0, R3, 340, 170, 0);
    crossj_kernel<<<blk(NB * 170), 256, 0, stream>>>(comb2, ja);
    wcat_kernel<<<blk(192L * 352), 256, 0, stream>>>(P(P_cl2Wr), P(P_cl2Wi), Wpad);
    gemm_raw(stream, comb2, 352, Wpad, 352, re, 192, 0, R3, 352, 170, 192, 4, 0);
    gemm_c(stream, comb2, 352, P(P_cl2D), Wpad, dcl, 192, 0, R3, 170, 170, 0);
    vn_apply_kernel<<<blk(NB * 170), 256, 0, stream>>>(
        re, 192, 0, dcl, 192, 0, 170, comb2, 352, 170,
        (const float*)nullptr, (const float*)nullptr, (const float*)nullptr,
        0.f, NB, 170);

    // ---- conv6 (D has 1 channel), writes into cols 0..340 of h682 ----
    gemm_c(stream, comb2, 352, P(P_c6W), Wpad, h682, 704, 0, R3, 340, 341, 0);
    gemm_c(stream, comb2, 352, P(P_c6D), Wpad, d6, 1, 0, R3, 340, 1, 0);
    bn_stats(stream, h682, 704, 0, statsb, NB, 341, 3);
    vn_apply_kernel<<<blk(NB * 341), 256, 0, stream>>>(
        h682, 704, 0, d6, 1, 0, 1, h682, 704, 0,
        statsb, P(P_c6g), P(P_c6b), 1.f / (float)NB, NB, 341);

    // ---- concat mean over N -> cols 341..681 ----
    meanfill_kernel<<<blk(Bb * 3 * 341), 256, 0, stream>>>(h682);

    // ---- VNStdFeature ----
    gemm_c(stream, h682, 704, P(P_s1W), Wpad, z1, 352, 0, R3, 682, 341, 0);
    gemm_c(stream, h682, 704, P(P_s1D), Wpad, zd, 352, 0, R3, 682, 341, 0);
    bn_stats(stream, z1, 352, 0, statsb, NB, 341, 3);
    vn_apply_kernel<<<blk(NB * 341), 256, 0, stream>>>(
        z1, 352, 0, zd, 352, 0, 341, z1, 352, 0,
        statsb, P(P_s1g), P(P_s1b), 1.f / (float)NB, NB, 341);
    gemm_c(stream, z1, 352, P(P_s2W), Wpad, z2, 192, 0, R3, 341, 170, 0);
    gemm_c(stream, z1, 352, P(P_s2D), Wpad, zd, 192, 0, R3, 341, 170, 0);
    bn_stats(stream, z2, 192, 0, statsb, NB, 170, 3);
    vn_apply_kernel<<<blk(NB * 170), 256, 0, stream>>>(
        z2, 192, 0, zd, 192, 0, 170, z2, 192, 0,
        statsb, P(P_s2g), P(P_s2b), 1.f / (float)NB, NB, 170);
    gemm_c(stream, z2, 192, P(P_s3W), Wpad, z0, 3, 0, R3, 170, 3, 0);

    // ---- frame projection, global pooling, label branch, assembly ----
    stdfeat_kernel<<<blk(NB * 682), 256, 0, stream>>>(h682, 704, 682, z0,
                                                      hstd, 2046, 0);
    colmax_kernel<<<blk(Bb * 2046), 256, 0, stream>>>(hstd, hg);
    label_kernel<<<1, 64, 0, stream>>>(lab, P(P_c7W), P(P_c7g), P(P_c7b), lf);
    fillg_kernel<<<blk(NB * 2110), 256, 0, stream>>>(Hc, hg, lf);
    stdfeat_kernel<<<blk(NB * 63), 256, 0, stream>>>(x123, 64, 63, z0,
                                                     Hc, 2304, 2110);

    // ---- head convs ----
    gemm_c(stream, Hc, 2304, P(P_c8W), Wpad, Y8, 256, 0, NB, 2299, 256, 0);
    bn_stats(stream, Y8, 256, 0, statsb, NB, 256, 1);
    bn_lrelu_kernel<<<blk(NB * 256), 256, 0, stream>>>(
        Y8, statsb, P(P_c8g), P(P_c8b), 1.f / (float)NB, NB * 256, 256);

    gemm_c(stream, Y8, 256, P(P_c9W), Wpad, Y9, 256, 0, NB, 256, 256, 0);
    bn_stats(stream, Y9, 256, 0, statsb, NB, 256, 1);
    bn_lrelu_kernel<<<blk(NB * 256), 256, 0, stream>>>(
        Y9, statsb, P(P_c9g), P(P_c9b), 1.f / (float)NB, NB * 256, 256);

    gemm_c(stream, Y9, 256, P(P_c10W), Wpad, Y10, 128, 0, NB, 256, 128, 0);
    bn_stats(stream, Y10, 128, 0, statsb, NB, 128, 1);
    bn_lrelu_kernel<<<blk(NB * 128), 256, 0, stream>>>(
        Y10, statsb, P(P_c10g), P(P_c10b), 1.f / (float)NB, NB * 128, 128);

    gemm_c(stream, Y10, 128, P(P_c11W), Wpad, outp, 50, 0, NB, 128, 50, 1);
}
